// MemoryAsContextTitan_53455162966286
// MI455X (gfx1250) — matrix operand order repaired and checked by reading, not compiled, not run on hardware
//
#include <hip/hip_runtime.h>

// ---------------- problem constants ----------------
#define B_    4
#define S_    3968
#define D_    512
#define H_    8
#define HD_   64
#define C_    496
#define NPM_  32
#define MEM_  1024
#define NCH_  8

typedef __attribute__((ext_vector_type(16))) __bf16 bfrag;
typedef __attribute__((ext_vector_type(8)))  float  v8f;

// ---------------- generic batched bf16 WMMA GEMM ----------------
// C[z][m][n] = scale * sum_k A[z][m][k] * B[z][n*bN + k]  (+ bias[n])
// A is k-contiguous (stride 1 along k, aR along m).
// B is k-contiguous (stride 1 along k, bN along n)  -- i.e. "B^T" storage.
// batch offset for tensor T: (z / nInner) * T_outer + (z % nInner) * T_inner
// fp32 out: row-major (cfR).  bf16 out: general strides cbM (along m), cbN (along n)
// so the same kernel can emit transposed bf16 activations (V^T) for downstream GEMMs.
struct GArg {
    const __bf16* A;   long long aO, aI; int aR;
    const __bf16* Bp;  long long bO, bI; int bN;
    const float*  bias;
    float*        Cf;  long long cfO, cfI; int cfR;
    __bf16*       Cb;  long long cbO, cbI; int cbM, cbN;
    int M, N, K, nInner;
    float scale;
};

__global__ __launch_bounds__(256)
void gemm_bf16_kernel(GArg g)
{
    const int bM  = blockIdx.x * 64;
    const int bN0 = blockIdx.y * 64;
    const long long zo = blockIdx.z / g.nInner, zi = blockIdx.z % g.nInner;

    const __bf16* A  = g.A  + zo * g.aO + zi * g.aI;
    const __bf16* Bp = g.Bp + zo * g.bO + zi * g.bI;

    __shared__ __align__(16) __bf16 lA [64][72];   // 64 m-rows x 64 k (+8 pad, keeps 16B alignment)
    __shared__ __align__(16) __bf16 lBt[64][72];   // 64 n-rows x 64 k

    const int tid  = threadIdx.x;
    const int lane = tid & 31;
    const int wave = tid >> 5;          // 8 waves: 2 along M x 4 along N
    const int wm   = (wave & 1) * 32;   // each wave: 32 rows (2 WMMA M-subtiles)
    const int wn   = (wave >> 1) * 16;  // 16 cols

    // staging: thread -> one row, 16 contiguous k (32B = 2x b128)
    const int rS = tid >> 2;
    const int c0 = (tid & 3) * 16;
    const bool inR = (bM + rS) < g.M;
    const __bf16* aPtr = A  + (long long)(bM  + rS) * g.aR + c0;
    const __bf16* bPtr = Bp + (long long)(bN0 + rS) * g.bN + c0;

    uint4 ra0, ra1, rb0, rb1;
    auto loadRegs = [&](int k0) {
        if (inR) {
            ra0 = *(const uint4*)(aPtr + k0);
            ra1 = *(const uint4*)(aPtr + k0 + 8);
        } else {
            ra0 = make_uint4(0u, 0u, 0u, 0u);
            ra1 = make_uint4(0u, 0u, 0u, 0u);
        }
        rb0 = *(const uint4*)(bPtr + k0);
        rb1 = *(const uint4*)(bPtr + k0 + 8);
    };
    auto storeLds = [&]() {
        *(uint4*)&lA [rS][c0]     = ra0;
        *(uint4*)&lA [rS][c0 + 8] = ra1;
        *(uint4*)&lBt[rS][c0]     = rb0;
        *(uint4*)&lBt[rS][c0 + 8] = rb1;
    };

    union FU { uint4 u[2]; bfrag f; };
    v8f acc0 = {0.f,0.f,0.f,0.f,0.f,0.f,0.f,0.f};
    v8f acc1 = {0.f,0.f,0.f,0.f,0.f,0.f,0.f,0.f};

    const int fm = lane & 15;           // fragment row/col within 16
    const int kh = (lane >> 4) * 8;     // A: lanes 0-15 K{0..7,16..23}; 16-31 K{8..15,24..31}
    const int kb = (lane >> 4) * 8;     // B: same K order as A

    auto computeTile = [&]() {
#pragma unroll
        for (int ks = 0; ks < 64; ks += 32) {
            FU bu, a0, a1;
            bu.u[0] = *(const uint4*)&lBt[wn + fm][ks + kb];
            bu.u[1] = *(const uint4*)&lBt[wn + fm][ks + kb + 16];
            a0.u[0] = *(const uint4*)&lA [wm + fm][ks + kh];
            a0.u[1] = *(const uint4*)&lA [wm + fm][ks + 16 + kh];
            a1.u[0] = *(const uint4*)&lA [wm + 16 + fm][ks + kh];
            a1.u[1] = *(const uint4*)&lA [wm + 16 + fm][ks + 16 + kh];
            acc0 = __builtin_amdgcn_wmma_f32_16x16x32_bf16(false, a0.f, false, bu.f,
                                                           (short)0, acc0, false, false);
            acc1 = __builtin_amdgcn_wmma_f32_16x16x32_bf16(false, a1.f, false, bu.f,
                                                           (short)0, acc1, false, false);
        }
    };

    // software pipeline: prefetch tile i+1 into regs while computing tile i from LDS
    loadRegs(0);
    storeLds();
    for (int k0 = 64; k0 < g.K; k0 += 64) {
        __syncthreads();
        loadRegs(k0);
        computeTile();
        __syncthreads();
        storeLds();
    }
    __syncthreads();
    computeTile();

    // writeback: VGPR j -> row (j + (lane>>4)*8), col (lane&15)
    const int n = bN0 + wn + fm;
    const long long cfo = zo * g.cfO + zi * g.cfI;
    const long long cbo = zo * g.cbO + zi * g.cbI;
    const float bias = g.bias ? g.bias[n] : 0.f;
#pragma unroll
    for (int t = 0; t < 2; ++t) {
        const v8f acc = t ? acc1 : acc0;
        const int mB = bM + wm + t * 16 + (lane >> 4) * 8;
#pragma unroll
        for (int j = 0; j < 8; ++j) {
            const int m = mB + j;
            if (m < g.M) {
                const float c = acc[j] * g.scale + bias;
                if (g.Cf) g.Cf[cfo + (long long)m * g.cfR + n] = c;
                if (g.Cb) g.Cb[cbo + (long long)m * g.cbM + (long long)n * g.cbN] = (__bf16)c;
            }
        }
    }
}

// ---------------- row softmax: fp32 scores -> bf16 probs ----------------
__global__ __launch_bounds__(256)
void softmax_rows(const float* __restrict__ Ss, __bf16* __restrict__ P, int ncol)
{
    const long long row = blockIdx.x;
    const float* s = Ss + row * ncol;
    __bf16*      p = P  + row * ncol;
    __shared__ float red[256];

    float mx = -3.4e38f;
    for (int i = threadIdx.x; i < ncol; i += 256) mx = fmaxf(mx, s[i]);
    red[threadIdx.x] = mx; __syncthreads();
    for (int o = 128; o; o >>= 1) {
        if (threadIdx.x < o) red[threadIdx.x] = fmaxf(red[threadIdx.x], red[threadIdx.x + o]);
        __syncthreads();
    }
    mx = red[0]; __syncthreads();

    float sum = 0.f;
    for (int i = threadIdx.x; i < ncol; i += 256) sum += __expf(s[i] - mx);
    red[threadIdx.x] = sum; __syncthreads();
    for (int o = 128; o; o >>= 1) {
        if (threadIdx.x < o) red[threadIdx.x] += red[threadIdx.x + o];
        __syncthreads();
    }
    const float inv = 1.f / red[0];
    for (int i = threadIdx.x; i < ncol; i += 256)
        p[i] = (__bf16)(__expf(s[i] - mx) * inv);
}

// ---------------- elementwise helpers ----------------
__global__ __launch_bounds__(256)
void cvt_w_transpose(const float* __restrict__ w, __bf16* __restrict__ o)
{   // w: [512][512] row-major (k,n) -> o: [n][k] (k-contiguous)
    const int i = blockIdx.x * 256 + threadIdx.x;   // output index
    if (i >= D_ * D_) return;
    const int k = i % D_, n = i / D_;
    o[i] = (__bf16)w[(long long)k * D_ + n];
}

__global__ __launch_bounds__(256)
void chunk_to_bf16(const float* __restrict__ x, __bf16* __restrict__ o, int ch)
{
    const int i = blockIdx.x * 256 + threadIdx.x;
    if (i >= B_ * C_ * D_) return;
    const int d = i % D_, r = (i / D_) % C_, b = i / (C_ * D_);
    o[i] = (__bf16)x[((long long)b * S_ + (long long)ch * C_ + r) * D_ + d];
}

__global__ __launch_bounds__(256)
void build_combined(const float* __restrict__ pm, const float* __restrict__ hist,
                    const float* __restrict__ x, __bf16* __restrict__ o, int ch)
{
    const int i = blockIdx.x * 256 + threadIdx.x;
    if (i >= B_ * MEM_ * D_) return;
    const int d = i % D_, r = (i / D_) % MEM_, b = i / (MEM_ * D_);
    float v;
    if (r < NPM_)           v = pm[r * D_ + d];
    else if (r < NPM_ + C_) v = hist[((long long)b * C_ + (r - NPM_)) * D_ + d];
    else                    v = x[((long long)b * S_ + (long long)ch * C_ + (r - NPM_ - C_)) * D_ + d];
    o[i] = (__bf16)v;
}

__global__ __launch_bounds__(256)
void ema_update(float* __restrict__ mem, const float* __restrict__ att, __bf16* __restrict__ membf)
{
    const int i = blockIdx.x * 256 + threadIdx.x;
    if (i >= B_ * MEM_ * D_) return;
    const float m = 0.9f * mem[i] + 0.1f * att[i];
    mem[i]   = m;
    membf[i] = (__bf16)m;
}

__global__ __launch_bounds__(256)
void finalize_out(const float* __restrict__ att, const float* __restrict__ mout,
                  float* __restrict__ out, int ch)
{
    const int i = blockIdx.x * 256 + threadIdx.x;
    if (i >= B_ * C_ * D_) return;
    const int d = i % D_, r = (i / D_) % C_, b = i / (C_ * D_);
    const long long src = ((long long)b * MEM_ + (MEM_ - C_) + r) * D_ + d;
    out[((long long)b * S_ + (long long)ch * C_ + r) * D_ + d] = att[src] * mout[src];
}

// ---------------- host-side GEMM launcher ----------------
static void launch_gemm(hipStream_t st,
                        const __bf16* A, long long aO, long long aI, int aR,
                        const __bf16* Bp, long long bO, long long bI, int bN,
                        const float* bias,
                        float* Cf, long long cfO, long long cfI, int cfR,
                        __bf16* Cb, long long cbO, long long cbI, int cbM, int cbN,
                        int M, int N, int K, int nBatch, int nInner, float scale)
{
    GArg g;
    g.A = A; g.aO = aO; g.aI = aI; g.aR = aR;
    g.Bp = Bp; g.bO = bO; g.bI = bI; g.bN = bN;
    g.bias = bias;
    g.Cf = Cf; g.cfO = cfO; g.cfI = cfI; g.cfR = cfR;
    g.Cb = Cb; g.cbO = cbO; g.cbI = cbI; g.cbM = cbM; g.cbN = cbN;
    g.M = M; g.N = N; g.K = K; g.nInner = nInner; g.scale = scale;
    dim3 grid((M + 63) / 64, N / 64, nBatch);
    gemm_bf16_kernel<<<grid, 256, 0, st>>>(g);
}

static inline int eb(long long n) { return (int)((n + 255) / 256); }

extern "C" void kernel_launch(void* const* d_in, const int* in_sizes, int n_in,
                              void* d_out, int out_size, void* d_ws, size_t ws_size,
                              hipStream_t stream)
{
    (void)in_sizes; (void)n_in; (void)out_size; (void)ws_size;
    const float* x    = (const float*)d_in[0];
    const float* pm   = (const float*)d_in[1];
    const float* wq_o = (const float*)d_in[2];  const float* bq_o = (const float*)d_in[3];
    const float* mk_w = (const float*)d_in[4];  const float* mk_b = (const float*)d_in[5];
    const float* mv_w = (const float*)d_in[6];  const float* mv_b = (const float*)d_in[7];
    const float* mq_w = (const float*)d_in[8];  const float* mq_b = (const float*)d_in[9];
    const float* aq_w = (const float*)d_in[10]; const float* aq_b = (const float*)d_in[11];
    const float* ak_w = (const float*)d_in[12]; const float* ak_b = (const float*)d_in[13];
    const float* av_w = (const float*)d_in[14]; const float* av_b = (const float*)d_in[15];
    const float* ao_w = (const float*)d_in[16]; const float* ao_b = (const float*)d_in[17];
    float* out = (float*)d_out;

    // ---------------- workspace carving ----------------
    size_t off = 0;
    auto carve = [&](size_t bytes) -> void* {
        void* p = (char*)d_ws + off;
        off = (off + bytes + 255) & ~(size_t)255;
        return p;
    };
    __bf16* wT[8];   // transposed bf16 weights: [n][k], k-contiguous
    for (int i = 0; i < 8; ++i) wT[i] = (__bf16*)carve((size_t)D_ * D_ * 2);
    float*  memf    = (float*)carve((size_t)B_ * MEM_ * D_ * 4);
    __bf16* membf   = (__bf16*)carve((size_t)B_ * MEM_ * D_ * 2);
    __bf16* chunkbf = (__bf16*)carve((size_t)B_ * C_ * D_ * 2);
    __bf16* querybf = (__bf16*)carve((size_t)B_ * C_ * D_ * 2);
    __bf16* qpbf    = (__bf16*)carve((size_t)B_ * MEM_ * D_ * 2);   // 1984 or 4096 rows
    __bf16* kbf     = (__bf16*)carve((size_t)B_ * MEM_ * D_ * 2);
    __bf16* vTbf    = (__bf16*)carve((size_t)B_ * MEM_ * D_ * 2);   // per-batch [D][MEM]
    float*  histf   = (float*)carve((size_t)B_ * C_ * D_ * 4);
    __bf16* combbf  = (__bf16*)carve((size_t)B_ * MEM_ * D_ * 2);
    __bf16* Qabf    = (__bf16*)carve((size_t)B_ * MEM_ * D_ * 2);
    __bf16* Kabf    = (__bf16*)carve((size_t)B_ * MEM_ * D_ * 2);
    __bf16* VaTbf   = (__bf16*)carve((size_t)B_ * MEM_ * D_ * 2);   // per-batch [D][MEM]
    __bf16* Oabf    = (__bf16*)carve((size_t)B_ * MEM_ * D_ * 2);
    float*  attf    = (float*)carve((size_t)B_ * MEM_ * D_ * 4);
    __bf16* attbf   = (__bf16*)carve((size_t)B_ * MEM_ * D_ * 2);
    float*  moutf   = (float*)carve((size_t)B_ * MEM_ * D_ * 4);
    float*  scores  = (float*)carve((size_t)B_ * H_ * MEM_ * MEM_ * 4);
    __bf16* probs   = (__bf16*)carve((size_t)B_ * H_ * MEM_ * MEM_ * 2);

    const float inv_sqrt_d  = 0.044194173824159223f;  // 1/sqrt(512)
    const float inv_sqrt_hd = 0.125f;                 // 1/sqrt(64)
    const long long MD = (long long)MEM_ * D_;
    const long long MM = (long long)MEM_ * MEM_;

    // ---------------- setup: transposed bf16 weights, zero memory state ----------------
    const float* wsrc[8] = { wq_o, mk_w, mv_w, mq_w, aq_w, ak_w, av_w, ao_w };
    for (int i = 0; i < 8; ++i)
        cvt_w_transpose<<<eb(D_ * D_), 256, 0, stream>>>(wsrc[i], wT[i]);
    hipMemsetAsync(memf,  0, (size_t)B_ * MEM_ * D_ * 4, stream);
    hipMemsetAsync(membf, 0, (size_t)B_ * MEM_ * D_ * 2, stream);

    const int nMD = B_ * MEM_ * D_;
    const int nCD = B_ * C_ * D_;

    for (int ch = 0; ch < NCH_; ++ch) {
        chunk_to_bf16<<<eb(nCD), 256, 0, stream>>>(x, chunkbf, ch);

        // query = chunk @ Wq + bq   (M = 1984)
        launch_gemm(stream, chunkbf, 0, 0, D_, wT[0], 0, 0, D_, bq_o,
                    nullptr, 0, 0, 0, querybf, 0, 0, D_, 1, B_ * C_, D_, D_, 1, 1, 1.f);
        // qp = query @ mq_w + mq_b
        launch_gemm(stream, querybf, 0, 0, D_, wT[3], 0, 0, D_, mq_b,
                    nullptr, 0, 0, 0, qpbf, 0, 0, D_, 1, B_ * C_, D_, D_, 1, 1, 1.f);
        // k = mem @ mk_w + mk_b   (M = 4096, row-major out)
        launch_gemm(stream, membf, 0, 0, D_, wT[1], 0, 0, D_, mk_b,
                    nullptr, 0, 0, 0, kbf, 0, 0, D_, 1, B_ * MEM_, D_, D_, 1, 1, 1.f);
        // vT = (mem @ mv_w + mv_b)^T  per batch: [D][MEM]
        launch_gemm(stream, membf, MD, 0, D_, wT[2], 0, 0, D_, mv_b,
                    nullptr, 0, 0, 0, vTbf, MD, 0, 1, MEM_, MEM_, D_, D_, B_, 1, 1.f);
        // sc1[b] = qp[b] @ k[b]^T / sqrt(D)
        launch_gemm(stream, qpbf, (long long)C_ * D_, 0, D_, kbf, MD, 0, D_, nullptr,
                    scores, (long long)C_ * MEM_, 0, MEM_, nullptr, 0, 0, 0, 1,
                    C_, MEM_, D_, B_, 1, inv_sqrt_d);
        softmax_rows<<<B_ * C_, 256, 0, stream>>>(scores, probs, MEM_);
        // hist[b] = P[b] @ v[b]   (B operand = vT, k-contiguous)
        launch_gemm(stream, probs, (long long)C_ * MEM_, 0, MEM_, vTbf, MD, 0, MEM_, nullptr,
                    histf, (long long)C_ * D_, 0, D_, nullptr, 0, 0, 0, 1,
                    C_, D_, MEM_, B_, 1, 1.f);

        build_combined<<<eb(nMD), 256, 0, stream>>>(pm, histf, x, combbf, ch);

        // MHA projections
        launch_gemm(stream, combbf, 0, 0, D_, wT[4], 0, 0, D_, aq_b,
                    nullptr, 0, 0, 0, Qabf, 0, 0, D_, 1, B_ * MEM_, D_, D_, 1, 1, 1.f);
        launch_gemm(stream, combbf, 0, 0, D_, wT[5], 0, 0, D_, ak_b,
                    nullptr, 0, 0, 0, Kabf, 0, 0, D_, 1, B_ * MEM_, D_, D_, 1, 1, 1.f);
        // VaT per batch: [D][MEM]
        launch_gemm(stream, combbf, MD, 0, D_, wT[6], 0, 0, D_, av_b,
                    nullptr, 0, 0, 0, VaTbf, MD, 0, 1, MEM_, MEM_, D_, D_, B_, 1, 1.f);
        // sc2[b,h] = Q[b,:,h,:] @ K[b,:,h,:]^T / 8   (batched B*H)
        launch_gemm(stream, Qabf, MD, HD_, D_, Kabf, MD, HD_, D_, nullptr,
                    scores, (long long)H_ * MM, MM, MEM_, nullptr, 0, 0, 0, 1,
                    MEM_, MEM_, HD_, B_ * H_, H_, inv_sqrt_hd);
        softmax_rows<<<B_ * H_ * MEM_, 256, 0, stream>>>(scores, probs, MEM_);
        // O[b,:,h,:] = P[b,h] @ V[b,:,h,:]   (B operand = VaT rows h*64..h*64+63)
        launch_gemm(stream, probs, (long long)H_ * MM, MM, MEM_,
                    VaTbf, MD, (long long)HD_ * MEM_, MEM_, nullptr,
                    nullptr, 0, 0, 0, Oabf, MD, HD_, D_, 1,
                    MEM_, HD_, MEM_, B_ * H_, H_, 1.f);
        // attended = O @ ao_w + ao_b  (fp32 + bf16)
        launch_gemm(stream, Oabf, 0, 0, D_, wT[7], 0, 0, D_, ao_b,
                    attf, 0, 0, D_, attbf, 0, 0, D_, 1, B_ * MEM_, D_, D_, 1, 1, 1.f);

        ema_update<<<eb(nMD), 256, 0, stream>>>(memf, attf, membf);

        // retrieve(mem2, attended)
        launch_gemm(stream, attbf, 0, 0, D_, wT[3], 0, 0, D_, mq_b,
                    nullptr, 0, 0, 0, qpbf, 0, 0, D_, 1, B_ * MEM_, D_, D_, 1, 1, 1.f);
        launch_gemm(stream, membf, 0, 0, D_, wT[1], 0, 0, D_, mk_b,
                    nullptr, 0, 0, 0, kbf, 0, 0, D_, 1, B_ * MEM_, D_, D_, 1, 1, 1.f);
        launch_gemm(stream, membf, MD, 0, D_, wT[2], 0, 0, D_, mv_b,
                    nullptr, 0, 0, 0, vTbf, MD, 0, 1, MEM_, MEM_, D_, D_, B_, 1, 1.f);
        launch_gemm(stream, qpbf, MD, 0, D_, kbf, MD, 0, D_, nullptr,
                    scores, MM, 0, MEM_, nullptr, 0, 0, 0, 1,
                    MEM_, MEM_, D_, B_, 1, inv_sqrt_d);
        softmax_rows<<<B_ * MEM_, 256, 0, stream>>>(scores, probs, MEM_);
        launch_gemm(stream, probs, MM, 0, MEM_, vTbf, MD, 0, MEM_, nullptr,
                    moutf, MD, 0, D_, nullptr, 0, 0, 0, 1,
                    MEM_, D_, MEM_, B_, 1, 1.f);

        finalize_out<<<eb(nCD), 256, 0, stream>>>(attf, moutf, out, ch);
    }
}
